// MyRnn_24378234372288
// MI455X (gfx1250) — matrix operand order repaired and checked by reading, not compiled, not run on hardware
//
#include <hip/hip_runtime.h>
#include <cstdint>
#include <cstddef>

typedef __bf16 bf16_t;
typedef bf16_t v16bf __attribute__((ext_vector_type(16)));
typedef bf16_t v8bf  __attribute__((ext_vector_type(8)));
typedef float  v8f   __attribute__((ext_vector_type(8)));

#define UNITS_C 1024
#define BATCH_C 1024
#define SEQ_C   80
#define EMB_C   100
#define EMBP_C  128   // EMB padded to a multiple of 32 (WMMA K-step) with zeros

// ---------------------------------------------------------------------------
// Fragment loaders matching CDNA5 WMMA VGPR layouts (CDNA5 ISA).
// ---------------------------------------------------------------------------

// A fragment: 16x32 bf16, row-major source [M x K] (ld = K stride).
// lane L in 0..15  : M=L,    VGPR0-3 = K 0..7,  VGPR4-7 = K 16..23
// lane L in 16..31 : M=L-16, VGPR0-3 = K 8..15, VGPR4-7 = K 24..31
__device__ __forceinline__ v16bf load_a_frag(const bf16_t* __restrict__ base,
                                             int row, int ld, int k, int lane) {
  const int r  = lane & 15;
  const int hs = lane >> 4;
  const bf16_t* p = base + (size_t)(row + r) * (size_t)ld + (size_t)(k + hs * 8);
  union { v16bf f; v8bf h[2]; } u;
  u.h[0] = *reinterpret_cast<const v8bf*>(p);        // 16B: K 0..7  (or 8..15)
  u.h[1] = *reinterpret_cast<const v8bf*>(p + 16);   // 16B: K 16..23 (or 24..31)
  return u.f;
}

// B fragment: 32x16 bf16 consumed from an N-major (pre-transposed) weight
// buffer WT[N][K] (ld = K stride).  Same K order as the A fragment:
// lane L in 0..15 holds N=L, K 0..7 then K 16..23; lane L+16 holds
// K 8..15 then K 24..31.  -> two 16B loads per lane.
__device__ __forceinline__ v16bf load_b_frag(const bf16_t* __restrict__ baseT,
                                             int col, int ld, int k, int lane) {
  const int r  = lane & 15;
  const int hs = lane >> 4;
  const bf16_t* p = baseT + (size_t)(col + r) * (size_t)ld + (size_t)(k + hs * 8);
  union { v16bf f; v8bf h[2]; } u;
  u.h[0] = *reinterpret_cast<const v8bf*>(p);
  u.h[1] = *reinterpret_cast<const v8bf*>(p + 16);
  return u.f;
}

// ---------------------------------------------------------------------------
// Fused RNN cell:  Hout = tanh( A[B x Ka] @ WxT^T  +  Hprev[B x U] @ WhT^T + b )
// WxT: [U x Ka] N-major, WhT: [U x U] N-major.  Both GEMMs accumulate into the
// same WMMA accumulators.  Block tile 64(M) x 64(N), 4 waves of 32x32.
// Grid: (U/64, B/64) = (16, 16) -> 256 workgroups (spread wide across WGPs:
// the 160-launch dependency chain is latency-bound, so keep per-wave path
// short and WGP coverage high; all operands are L2-resident).
// ---------------------------------------------------------------------------
__global__ __launch_bounds__(128)
void rnn_cell_wmma(const bf16_t* __restrict__ A, int Ka,
                   const bf16_t* __restrict__ WxT,
                   const bf16_t* __restrict__ Hprev,
                   const bf16_t* __restrict__ WhT,
                   const float*  __restrict__ bias,
                   bf16_t* __restrict__ Hout) {
  const int lane = threadIdx.x & 31;
  const int wave = threadIdx.x >> 5;
  const int wm   = wave >> 1;          // 0..1 -> 32-row group
  const int wn   = wave & 1;           // 0..1 -> 32-col group
  const int mBase = blockIdx.y * 64 + wm * 32;
  const int nBase = blockIdx.x * 64 + wn * 32;

  v8f acc[2][2];
  const v8f zf = {0.f, 0.f, 0.f, 0.f, 0.f, 0.f, 0.f, 0.f};
#pragma unroll
  for (int i = 0; i < 2; ++i)
#pragma unroll
    for (int j = 0; j < 2; ++j) acc[i][j] = zf;

  // ---- GEMM 1: A @ Wx  (Ka = 128 or 1024) ---------------------------------
#pragma unroll 2
  for (int k = 0; k < Ka; k += 32) {
    v16bf a[2], b[2];
#pragma unroll
    for (int i = 0; i < 2; ++i) a[i] = load_a_frag(A,   mBase + i * 16, Ka, k, lane);
#pragma unroll
    for (int j = 0; j < 2; ++j) b[j] = load_b_frag(WxT, nBase + j * 16, Ka, k, lane);
#pragma unroll
    for (int i = 0; i < 2; ++i)
#pragma unroll
      for (int j = 0; j < 2; ++j)
        acc[i][j] = __builtin_amdgcn_wmma_f32_16x16x32_bf16(
            false, a[i], false, b[j], (short)0, acc[i][j], false, false);
  }

  // ---- GEMM 2: Hprev @ Wh (K = UNITS) -------------------------------------
#pragma unroll 2
  for (int k = 0; k < UNITS_C; k += 32) {
    // stream next chunk through GL2 (global_prefetch_b8)
    __builtin_prefetch((const void*)(Hprev + (size_t)(mBase + (lane & 15)) * UNITS_C + k + 64), 0, 1);
    __builtin_prefetch((const void*)(WhT   + (size_t)(nBase + (lane & 15)) * UNITS_C + k + 64), 0, 1);
    v16bf a[2], b[2];
#pragma unroll
    for (int i = 0; i < 2; ++i) a[i] = load_a_frag(Hprev, mBase + i * 16, UNITS_C, k, lane);
#pragma unroll
    for (int j = 0; j < 2; ++j) b[j] = load_b_frag(WhT,   nBase + j * 16, UNITS_C, k, lane);
#pragma unroll
    for (int i = 0; i < 2; ++i)
#pragma unroll
      for (int j = 0; j < 2; ++j)
        acc[i][j] = __builtin_amdgcn_wmma_f32_16x16x32_bf16(
            false, a[i], false, b[j], (short)0, acc[i][j], false, false);
  }

  // ---- epilogue: bias + tanh + bf16 store ---------------------------------
  // C/D layout: lane 0..15 -> N=lane, M=v; lane 16..31 -> N=lane-16, M=8+v
  const int r  = lane & 15;
  const int hs = lane >> 4;
#pragma unroll
  for (int j = 0; j < 2; ++j) {
    const int n  = nBase + j * 16 + r;
    const float bv = bias[n];
#pragma unroll
    for (int i = 0; i < 2; ++i) {
#pragma unroll
      for (int v = 0; v < 8; ++v) {
        const int m = mBase + i * 16 + hs * 8 + v;
        const float f = tanhf(acc[i][j][v] + bv);
        Hout[(size_t)m * UNITS_C + n] = (bf16_t)f;
      }
    }
  }
}

// ---------------------------------------------------------------------------
// Setup kernels (run once per launch; all scratch recomputed deterministically)
// ---------------------------------------------------------------------------

// dst[n*Kpad + k] = bf16(src[k*N + n]) for k < K, else 0   (N-major weights)
__global__ void transpose_to_bf16(const float* __restrict__ src, bf16_t* __restrict__ dst,
                                  int K, int N, int Kpad) {
  const int idx = blockIdx.x * 256 + threadIdx.x;
  if (idx >= N * Kpad) return;
  const int n = idx / Kpad;
  const int k = idx - n * Kpad;
  const float v = (k < K) ? src[(size_t)k * N + n] : 0.f;
  dst[idx] = (bf16_t)v;
}

// Xe[(t*B + b)*128 + k] = bf16(emb[x[b,t]][k])  (k<100; zero-padded to 128)
__global__ void embed_to_bf16(const int* __restrict__ x, const float* __restrict__ emb,
                              bf16_t* __restrict__ Xe) {
  const int gid = blockIdx.x * 2 + (threadIdx.x >> 7);   // row = t*B + b
  const int k   = threadIdx.x & 127;
  const int t   = gid / BATCH_C;
  const int b   = gid - t * BATCH_C;
  const int tok = x[b * SEQ_C + t];
  const float v = (k < EMB_C) ? emb[(size_t)tok * EMB_C + k] : 0.f;
  Xe[(size_t)gid * EMBP_C + k] = (bf16_t)v;
}

__global__ void zero_u32(uint32_t* __restrict__ p, int n) {
  const int i = blockIdx.x * 256 + threadIdx.x;
  if (i < n) p[i] = 0u;
}

// out[b] = sigmoid( sum_u h1[b][u] * Wout[u] + bout )
__global__ __launch_bounds__(256)
void dense_sigmoid(const bf16_t* __restrict__ h1, const float* __restrict__ Wout,
                   const float* __restrict__ bout, float* __restrict__ out) {
  const int b = blockIdx.x;
  const int tid = threadIdx.x;
  float s = 0.f;
  for (int u = tid; u < UNITS_C; u += 256)
    s += (float)h1[(size_t)b * UNITS_C + u] * Wout[u];
  __shared__ float red[256];
  red[tid] = s;
  __syncthreads();
  for (int off = 128; off > 0; off >>= 1) {
    if (tid < off) red[tid] += red[tid + off];
    __syncthreads();
  }
  if (tid == 0) {
    const float z = red[0] + bout[0];
    out[b] = 1.f / (1.f + expf(-z));
  }
}

// ---------------------------------------------------------------------------
// Host orchestration
// ---------------------------------------------------------------------------
extern "C" void kernel_launch(void* const* d_in, const int* in_sizes, int n_in,
                              void* d_out, int out_size, void* d_ws, size_t ws_size,
                              hipStream_t stream) {
  (void)in_sizes; (void)n_in; (void)out_size; (void)ws_size;

  const int*   x    = (const int*)  d_in[0];
  const float* emb  = (const float*)d_in[1];
  const float* W0x  = (const float*)d_in[2];
  const float* W0h  = (const float*)d_in[3];
  const float* b0   = (const float*)d_in[4];
  const float* W1x  = (const float*)d_in[5];
  const float* W1h  = (const float*)d_in[6];
  const float* b1   = (const float*)d_in[7];
  const float* Wout = (const float*)d_in[8];
  const float* bout = (const float*)d_in[9];
  float* out = (float*)d_out;

  // ---- workspace layout (256B aligned), ~34.5 MB total ----
  char* ws = (char*)d_ws;
  size_t off = 0;
  auto carve = [&](size_t bytes) -> void* {
    void* p = ws + off;
    off = (off + bytes + 255) & ~(size_t)255;
    return p;
  };
  bf16_t* W0xT = (bf16_t*)carve((size_t)UNITS_C * EMBP_C  * 2);
  bf16_t* W0hT = (bf16_t*)carve((size_t)UNITS_C * UNITS_C * 2);
  bf16_t* W1xT = (bf16_t*)carve((size_t)UNITS_C * UNITS_C * 2);
  bf16_t* W1hT = (bf16_t*)carve((size_t)UNITS_C * UNITS_C * 2);
  bf16_t* Xe   = (bf16_t*)carve((size_t)SEQ_C * BATCH_C * EMBP_C * 2);
  bf16_t* Hbuf = (bf16_t*)carve((size_t)4 * BATCH_C * UNITS_C * 2);  // h0[2], h1[2]
  bf16_t* h0b[2] = { Hbuf,                                 Hbuf + (size_t)1 * BATCH_C * UNITS_C };
  bf16_t* h1b[2] = { Hbuf + (size_t)2 * BATCH_C * UNITS_C, Hbuf + (size_t)3 * BATCH_C * UNITS_C };

  // ---- setup: weight layout conversion, embedding gather, state init ----
  transpose_to_bf16<<<(UNITS_C * EMBP_C  + 255) / 256, 256, 0, stream>>>(W0x, W0xT, EMB_C,   UNITS_C, EMBP_C);
  transpose_to_bf16<<<(UNITS_C * UNITS_C + 255) / 256, 256, 0, stream>>>(W0h, W0hT, UNITS_C, UNITS_C, UNITS_C);
  transpose_to_bf16<<<(UNITS_C * UNITS_C + 255) / 256, 256, 0, stream>>>(W1x, W1xT, UNITS_C, UNITS_C, UNITS_C);
  transpose_to_bf16<<<(UNITS_C * UNITS_C + 255) / 256, 256, 0, stream>>>(W1h, W1hT, UNITS_C, UNITS_C, UNITS_C);
  embed_to_bf16<<<SEQ_C * BATCH_C / 2, 256, 0, stream>>>(x, emb, Xe);
  {
    const int n32 = (int)((size_t)4 * BATCH_C * UNITS_C * 2 / 4);  // 8MB of zeros
    zero_u32<<<(n32 + 255) / 256, 256, 0, stream>>>((uint32_t*)Hbuf, n32);
  }

  // ---- 80 sequential timesteps x 2 layers: 160 dependent WMMA GEMMs ----
  const dim3 grid(UNITS_C / 64, BATCH_C / 64);  // (16, 16) = 256 WGs
  for (int t = 0; t < SEQ_C; ++t) {
    const int cur = t & 1, nxt = (t + 1) & 1;
    rnn_cell_wmma<<<grid, 128, 0, stream>>>(
        Xe + (size_t)t * BATCH_C * EMBP_C, EMBP_C, W0xT, h0b[cur], W0hT, b0, h0b[nxt]);
    rnn_cell_wmma<<<grid, 128, 0, stream>>>(
        h0b[nxt], UNITS_C, W1xT, h1b[cur], W1hT, b1, h1b[nxt]);
  }

  // final h1 lives in buffer (SEQ_C & 1) == 0
  dense_sigmoid<<<BATCH_C, 256, 0, stream>>>(h1b[SEQ_C & 1], Wout, bout, out);
}
